// ReprogrammingLayerSample_72499047957243
// MI455X (gfx1250) — matrix operand order repaired and checked by reading, not compiled, not run on hardware
//
#include <hip/hip_runtime.h>
#include <hip/hip_bf16.h>

// ---------------------------------------------------------------------------
// ReprogrammingLayer for MI455X (gfx1250): all contractions via
// v_wmma_f32_16x16x32_bf16 (fp32 accumulate). 128x128 double-buffered WMMA
// GEMM (8 WMMA / wave / K-step), fused attention with 280KB dynamic LDS
// (CDNA5 WGP has 320KB), fp32 global atomics for the head-mean.
// ---------------------------------------------------------------------------

typedef __attribute__((ext_vector_type(16))) __bf16 v16bf;
typedef __attribute__((ext_vector_type(8)))  __bf16 v8bf;
typedef __attribute__((ext_vector_type(8)))  float  v8f;

static constexpr int Bn = 4, Ln = 1024, Sn = 1024, Hn = 16, En = 64;
static constexpr int DM = 1024, DLLM = 4096, HE = 1024;

__device__ __forceinline__ v8f wmma_bf16(v16bf a, v16bf b, v8f c) {
  // (neg_a, A, neg_b, B, c_mod, C, reuse_a, reuse_b)
  return __builtin_amdgcn_wmma_f32_16x16x32_bf16(false, a, false, b, (short)0, c,
                                                 false, false);
}

__device__ __forceinline__ v16bf cat8(v8bf lo, v8bf hi) {
  return __builtin_shufflevector(lo, hi, 0, 1, 2, 3, 4, 5, 6, 7,
                                 8, 9, 10, 11, 12, 13, 14, 15);
}

__device__ __forceinline__ float leaky(float x) {
  return (x < 0.f) ? 0.01f * x : x;
}

// ------------------------------ elementwise --------------------------------

__global__ __launch_bounds__(256) void cast_f32_bf16(const float* __restrict__ s,
                                                     __bf16* __restrict__ d, int n) {
  int i = blockIdx.x * 256 + threadIdx.x;
  if (i < n) d[i] = (__bf16)s[i];
}

__global__ __launch_bounds__(256) void zero_f32(float* __restrict__ p, int n) {
  int i = blockIdx.x * 256 + threadIdx.x;
  if (i < n) p[i] = 0.f;
}

// Td[b][h][l][e] = bf16( T[b,l,h*64+e] * diag(Ws1)[e] * diag(Ws2)[e] )
__global__ __launch_bounds__(256) void pack_T(const float* __restrict__ Tf,
                                              const float* __restrict__ Ws1,
                                              const float* __restrict__ Ws2,
                                              __bf16* __restrict__ Td, int n) {
  int i = blockIdx.x * 256 + threadIdx.x;
  if (i >= n) return;
  int e = i & 63, l = (i >> 6) & 1023, h = (i >> 16) & 15, b = i >> 20;
  float d = Ws1[e * 64 + e] * Ws2[e * 64 + e];
  Td[i] = (__bf16)(Tf[((size_t)(b * Ln + l)) * HE + h * 64 + e] * d);
}

// dst[h][s][e] = bf16( src[s, h*64+e] )
__global__ __launch_bounds__(256) void pack_hse(const float* __restrict__ s,
                                                __bf16* __restrict__ d, int n) {
  int i = blockIdx.x * 256 + threadIdx.x;
  if (i >= n) return;
  int e = i & 63, sr = (i >> 6) & 1023, h = i >> 16;
  d[i] = (__bf16)s[(size_t)sr * HE + h * 64 + e];
}

// ------------------------------ generic GEMM -------------------------------
// C[M,N] = A[M,K](bf16,row) * B[K,N](bf16,row) + bias[N]
// 128x128 block tile, K-step 32, double-buffered LDS, 8 waves (4x2),
// each wave owns a 32x64 region = 2x4 WMMA accumulators.
// Requires M%128==0, N%128==0, K%32==0.
__global__ __launch_bounds__(256) void gemm_bf16(const __bf16* __restrict__ A,
                                                 const __bf16* __restrict__ B,
                                                 const float* __restrict__ bias,
                                                 float* __restrict__ C,
                                                 int M, int N, int K) {
  __shared__ __bf16 As[2][128][48];   // 12KB per buffer (96B pitch, 16B aligned)
  __shared__ __bf16 Bs[2][32][136];   // 8.5KB per buffer (272B pitch)
  const int tid = threadIdx.x, lane = tid & 31, w = tid >> 5;
  const int mbase = blockIdx.y * 128, nbase = blockIdx.x * 128;
  const int mt0 = (w & 3) * 2;   // wave's first M tile (of 8)
  const int nt0 = (w >> 2) * 4;  // wave's first N tile (of 8)
  // staging assignments
  const int ar = tid >> 1, ac = (tid & 1) * 16;  // A: 128 rows x 32 (16 halfs/thr)
  const int br = tid >> 3, bc = (tid & 7) * 16;  // B: 32 rows x 128 (16 halfs/thr)
  const __bf16* agp = &A[(size_t)(mbase + ar) * K + ac];
  const __bf16* bgp = &B[(size_t)br * N + nbase + bc];

  v8f acc[2][4];
#pragma unroll
  for (int i = 0; i < 2; ++i)
#pragma unroll
    for (int j = 0; j < 4; ++j) acc[i][j] = v8f{0, 0, 0, 0, 0, 0, 0, 0};

  // preload tile 0
  *(v8bf*)&As[0][ar][ac]     = *(const v8bf*)(agp);
  *(v8bf*)&As[0][ar][ac + 8] = *(const v8bf*)(agp + 8);
  *(v8bf*)&Bs[0][br][bc]     = *(const v8bf*)(bgp);
  *(v8bf*)&Bs[0][br][bc + 8] = *(const v8bf*)(bgp + 8);
  __syncthreads();

  const int frow = (lane & 15);
  const int fkc = (lane >> 4) * 8;
  int cur = 0;
  for (int kb = 0; kb < K; kb += 32) {
    if (kb + 32 < K) {
      // prefetch two steps ahead (global_prefetch_b8)
      if (kb + 64 < K) {
        __builtin_prefetch(agp + (size_t)(kb + 64), 0, 3);
        __builtin_prefetch(bgp + (size_t)(kb + 64) * N, 0, 3);
      }
      // stage next tile into the other buffer (no barrier needed: disjoint buf)
      const __bf16* ag = agp + (size_t)(kb + 32);
      const __bf16* bg = bgp + (size_t)(kb + 32) * N;
      v8bf a0 = *(const v8bf*)(ag);
      v8bf a1 = *(const v8bf*)(ag + 8);
      v8bf b0 = *(const v8bf*)(bg);
      v8bf b1 = *(const v8bf*)(bg + 8);
      *(v8bf*)&As[cur ^ 1][ar][ac]     = a0;
      *(v8bf*)&As[cur ^ 1][ar][ac + 8] = a1;
      *(v8bf*)&Bs[cur ^ 1][br][bc]     = b0;
      *(v8bf*)&Bs[cur ^ 1][br][bc + 8] = b1;
    }
    // compute on current buffer: 2 A frags, 4 B frags, 8 WMMA
    v16bf af[2];
#pragma unroll
    for (int i = 0; i < 2; ++i) {
      const __bf16* ap = &As[cur][(mt0 + i) * 16 + frow][fkc];
      af[i] = cat8(*(const v8bf*)ap, *(const v8bf*)(ap + 16));
    }
#pragma unroll
    for (int j = 0; j < 4; ++j) {
      const __bf16* bp = &Bs[cur][fkc][(nt0 + j) * 16 + frow];
      v16bf bfr;
#pragma unroll
      for (int s = 0; s < 8; ++s) {
        bfr[s] = bp[s * 136];
        bfr[8 + s] = bp[(16 + s) * 136];
      }
#pragma unroll
      for (int i = 0; i < 2; ++i) acc[i][j] = wmma_bf16(af[i], bfr, acc[i][j]);
    }
    __syncthreads();
    cur ^= 1;
  }

  const int rsub = 8 * (lane >> 4);
  const int cl = lane & 15;
#pragma unroll
  for (int i = 0; i < 2; ++i)
#pragma unroll
    for (int r = 0; r < 8; ++r) {
      int rowg = mbase + (mt0 + i) * 16 + rsub + r;
#pragma unroll
      for (int j = 0; j < 4; ++j) {
        int c = nbase + (nt0 + j) * 16 + cl;
        C[(size_t)rowg * N + c] = acc[i][j][r] + bias[c];
      }
    }
}

// ------------------------------ fused attention ----------------------------
// grid (L/64, H, B), 256 threads. Dynamic LDS:
//   sS  [64][1032] f32  (scores -> softmax A)          264192 B
//   ktr [64][136]  bf16 (K^T staging, 128-col s chunk)  17408 B
//   red [64][4]    f32  (row reductions)                 1024 B
static constexpr int SS_PITCH = 1032;
static constexpr int KTR_PITCH = 136;
static constexpr size_t ATTN_LDS =
    (size_t)64 * SS_PITCH * 4 + (size_t)64 * KTR_PITCH * 2 + 64 * 4 * 4;

__global__ __launch_bounds__(256) void attn_fused(
    const __bf16* __restrict__ Td,   // [B][H][L][E]  (T * d12)
    const __bf16* __restrict__ Kd,   // [H][S][E]
    const __bf16* __restrict__ Vd,   // [H][S][E]
    const float*  __restrict__ Tf,   // [B*L][HE]  (fp32 T, for rt == T)
    const float*  __restrict__ cw1,  // [E][H]
    const float*  __restrict__ cw2,  // [E][H]
    float* __restrict__ Amean,       // [B][L][S] (atomic accumulate)
    __bf16* __restrict__ repb) {     // [B*L][HE]
  extern __shared__ char smem[];
  float*  sS  = (float*)smem;
  __bf16* ktr = (__bf16*)(smem + (size_t)64 * SS_PITCH * 4);
  float*  red = (float*)(smem + (size_t)64 * SS_PITCH * 4 + (size_t)64 * KTR_PITCH * 2);

  const int tid = threadIdx.x, lane = tid & 31, w = tid >> 5;
  const int lt = blockIdx.x, h = blockIdx.y, b = blockIdx.z;
  const int lbase = lt * 64;
  const float scale = 0.125f;  // 1/sqrt(E)

  const __bf16* TdH = Td + (((size_t)b * Hn + h) * Ln + lbase) * En;
  const __bf16* KdH = Kd + (size_t)h * Sn * En;
  const __bf16* VdH = Vd + (size_t)h * Sn * En;

  // ---- phase 1: scores[64][1024] = scale * leaky( (T*d12) @ K^T ) ----
  {
    const int mt = w >> 1;            // 0..3
    const int ntb4 = (w & 1) * 4;     // 4 of 8 N tiles per wave
    // A fragments are invariant across s-chunks: hoist
    v16bf afk[2];
    {
      int row = mt * 16 + (lane & 15);
      int kcb = (lane >> 4) * 8;
#pragma unroll
      for (int ks = 0; ks < 2; ++ks) {
        const __bf16* ap = &TdH[(size_t)row * En + ks * 32 + kcb];
        afk[ks] = cat8(*(const v8bf*)ap, *(const v8bf*)(ap + 16));
      }
    }
    for (int sc = 0; sc < Sn; sc += 128) {
      {  // stage ktr[e][j] = K[h][sc+j][e]  (transpose in LDS)
        int s = tid >> 1;
        int e0 = (tid & 1) * 32;
        const __bf16* sp = &KdH[(size_t)(sc + s) * En + e0];
        v8bf q0 = *(const v8bf*)(sp);
        v8bf q1 = *(const v8bf*)(sp + 8);
        v8bf q2 = *(const v8bf*)(sp + 16);
        v8bf q3 = *(const v8bf*)(sp + 24);
#pragma unroll
        for (int j = 0; j < 8; ++j) {
          ktr[(e0 + j) * KTR_PITCH + s]      = q0[j];
          ktr[(e0 + 8 + j) * KTR_PITCH + s]  = q1[j];
          ktr[(e0 + 16 + j) * KTR_PITCH + s] = q2[j];
          ktr[(e0 + 24 + j) * KTR_PITCH + s] = q3[j];
        }
      }
      __syncthreads();
      v8f acc[4];
#pragma unroll
      for (int nt = 0; nt < 4; ++nt) acc[nt] = v8f{0, 0, 0, 0, 0, 0, 0, 0};
#pragma unroll
      for (int ks = 0; ks < 2; ++ks) {
        int ke = ks * 32 + (lane >> 4) * 8;
#pragma unroll
        for (int nt = 0; nt < 4; ++nt) {
          const __bf16* bp = &ktr[ke * KTR_PITCH + (ntb4 + nt) * 16 + (lane & 15)];
          v16bf bfr;
#pragma unroll
          for (int s = 0; s < 8; ++s) {
            bfr[s] = bp[s * KTR_PITCH];
            bfr[8 + s] = bp[(16 + s) * KTR_PITCH];
          }
          acc[nt] = wmma_bf16(afk[ks], bfr, acc[nt]);
        }
      }
      const int rbase = mt * 16 + 8 * (lane >> 4);
      const int cl = lane & 15;
#pragma unroll
      for (int nt = 0; nt < 4; ++nt)
#pragma unroll
        for (int r = 0; r < 8; ++r)
          sS[(rbase + r) * SS_PITCH + sc + (ntb4 + nt) * 16 + cl] =
              leaky(acc[nt][r]) * scale;
      __syncthreads();
    }
  }

  // ---- phase 2: softmax over S, accumulate A/H into Amean ----
  {
    const int row = tid >> 2, part = tid & 3;
    float* rowp = &sS[row * SS_PITCH];
    const int s0 = part * 256;
    float m = -1e30f;
    for (int j = 0; j < 256; ++j) m = fmaxf(m, rowp[s0 + j]);
    red[row * 4 + part] = m;
    __syncthreads();
    float rm = fmaxf(fmaxf(red[row * 4 + 0], red[row * 4 + 1]),
                     fmaxf(red[row * 4 + 2], red[row * 4 + 3]));
    float lsum = 0.f;
    for (int j = 0; j < 256; ++j) {
      float e = __expf(rowp[s0 + j] - rm);
      rowp[s0 + j] = e;
      lsum += e;
    }
    __syncthreads();
    red[row * 4 + part] = lsum;
    __syncthreads();
    float inv = 1.f / (red[row * 4 + 0] + red[row * 4 + 1] +
                       red[row * 4 + 2] + red[row * 4 + 3]);
    float* am = &Amean[((size_t)b * Ln + lbase + row) * Sn];
    for (int j = 0; j < 256; ++j) {
      float a = rowp[s0 + j] * inv;
      rowp[s0 + j] = a;
      atomicAdd(&am[s0 + j], a * (1.f / Hn));
    }
  }
  __syncthreads();

  // ---- phase 3: rv = A @ V; rep = leaky(rv * T * comb) -> bf16 ----
  {
    const int mt = w >> 1;
    const int ntb2 = (w & 1) * 2;   // 2 of 4 E tiles per wave
    v8f a0 = {0, 0, 0, 0, 0, 0, 0, 0};
    v8f a1 = {0, 0, 0, 0, 0, 0, 0, 0};
    for (int ks = 0; ks < 32; ++ks) {
      int row = mt * 16 + (lane & 15);
      int kc = ks * 32 + (lane >> 4) * 8;
      const float* sp = &sS[row * SS_PITCH + kc];
      v16bf af;
#pragma unroll
      for (int j = 0; j < 8; ++j) {
        af[j] = (__bf16)sp[j];
        af[8 + j] = (__bf16)sp[16 + j];
      }
      const __bf16* vp = &VdH[(size_t)(ks * 32 + (lane >> 4) * 8) * En + ntb2 * 16 + (lane & 15)];
      v16bf bf0, bf1;
#pragma unroll
      for (int j = 0; j < 8; ++j) {
        bf0[j] = vp[j * En];
        bf0[8 + j] = vp[(16 + j) * En];
        bf1[j] = vp[j * En + 16];
        bf1[8 + j] = vp[(16 + j) * En + 16];
      }
      a0 = wmma_bf16(af, bf0, a0);
      a1 = wmma_bf16(af, bf1, a1);
    }
    const int rbase = mt * 16 + 8 * (lane >> 4);
    const int cl = lane & 15;
#pragma unroll
    for (int r = 0; r < 8; ++r) {
      int l = lbase + rbase + r;
#pragma unroll
      for (int i = 0; i < 2; ++i) {
        int e = (ntb2 + i) * 16 + cl;
        float rv = i ? a1[r] : a0[r];
        float t = Tf[((size_t)b * Ln + l) * HE + h * 64 + e];
        float c = cw1[e * Hn + h] * cw2[e * Hn + h];
        repb[((size_t)b * Ln + l) * HE + h * 64 + e] = (__bf16)leaky(rv * t * c);
      }
    }
  }
}

// ------------------------------ argmax/argmin + gather ---------------------

__global__ __launch_bounds__(256) void argmm_kernel(const float* __restrict__ Am,
                                                    int* __restrict__ mxi,
                                                    int* __restrict__ mni, int rows) {
  int r = blockIdx.x * 256 + threadIdx.x;
  if (r >= rows) return;
  const float* p = Am + (size_t)r * Sn;
  float mx = p[0], mn = p[0];
  int xi = 0, ni = 0;
  for (int j = 1; j < Sn; ++j) {
    float v = p[j];
    if (v > mx) { mx = v; xi = j; }
    if (v < mn) { mn = v; ni = j; }
  }
  mxi[r] = xi;
  mni[r] = ni;
}

__global__ __launch_bounds__(256) void gather_kernel(const float* __restrict__ Kf,
                                                     const int* __restrict__ mxi,
                                                     const int* __restrict__ mni,
                                                     __bf16* __restrict__ gpos,
                                                     __bf16* __restrict__ gneg) {
  int row = blockIdx.x;
  int xi = mxi[row], ni = mni[row];
  for (int c = threadIdx.x; c < HE; c += 256) {
    gpos[(size_t)row * HE + c] = (__bf16)Kf[(size_t)xi * HE + c];
    gneg[(size_t)row * HE + c] = (__bf16)Kf[(size_t)ni * HE + c];
  }
}

// ------------------------------ launch -------------------------------------

extern "C" void kernel_launch(void* const* d_in, const int* in_sizes, int n_in,
                              void* d_out, int out_size, void* d_ws, size_t ws_size,
                              hipStream_t stream) {
  const float* tgt = (const float*)d_in[0];
  const float* src = (const float*)d_in[1];
  const float* val = (const float*)d_in[2];
  const float* Wq = (const float*)d_in[3];  const float* bq = (const float*)d_in[4];
  const float* Wk = (const float*)d_in[5];  const float* bk = (const float*)d_in[6];
  const float* Wv = (const float*)d_in[7];  const float* bv = (const float*)d_in[8];
  const float* Wo = (const float*)d_in[9];  const float* bo = (const float*)d_in[10];
  const float* Wsm = (const float*)d_in[11]; const float* bsv = (const float*)d_in[12];
  const float* Ws1 = (const float*)d_in[13]; const float* Ws2 = (const float*)d_in[14];
  const float* cw1 = (const float*)d_in[15]; const float* cw2 = (const float*)d_in[16];
  float* outp = (float*)d_out;

  uint8_t* ws = (uint8_t*)d_ws;
  size_t off = 0;
  auto take = [&](size_t bytes) {
    uint8_t* p = ws + off;
    off = (off + bytes + 255) & ~(size_t)255;
    return p;
  };
  const size_t NBL = (size_t)Bn * Ln;          // 4096 rows
  float*  Tf32  = (float*)take(NBL * HE * 4);                 // 16 MB
  float*  Kf32  = (float*)take((size_t)Sn * HE * 4);          // 4 MB
  float*  Vtmp  = (float*)take((size_t)Sn * HE * 4);          // 4 MB
  __bf16* Td    = (__bf16*)take((size_t)Bn * Hn * Ln * En * 2);
  __bf16* Kd    = (__bf16*)take((size_t)Hn * Sn * En * 2);
  __bf16* Vd    = (__bf16*)take((size_t)Hn * Sn * En * 2);
  float*  Amean = (float*)take((size_t)Bn * Ln * Sn * 4);     // 16 MB
  int*    mxi   = (int*)take(NBL * 4);
  int*    mni   = (int*)take(NBL * 4);
  __bf16* tgt16 = (__bf16*)take(NBL * DM * 2);   // reused as repb after T-GEMM
  __bf16* src16 = (__bf16*)take((size_t)Sn * DLLM * 2);  // reused as gpos
  __bf16* val16 = (__bf16*)take((size_t)Sn * DLLM * 2);  // reused as gneg
  __bf16* Wq16  = (__bf16*)take((size_t)DM * HE * 2);
  __bf16* Wk16  = (__bf16*)take((size_t)DLLM * HE * 2);
  __bf16* Wv16  = (__bf16*)take((size_t)DLLM * HE * 2);
  __bf16* Wo16  = (__bf16*)take((size_t)HE * DLLM * 2);
  __bf16* Ws16  = (__bf16*)take((size_t)HE * DLLM * 2);
  __bf16* repb  = tgt16;
  __bf16* gpos  = src16;
  __bf16* gneg  = val16;

  const dim3 blk(256);
  auto nb = [](int n) { return dim3((n + 255) / 256); };

  // 1) bf16 casts
  cast_f32_bf16<<<nb(4194304), blk, 0, stream>>>(tgt, tgt16, 4194304);
  cast_f32_bf16<<<nb(4194304), blk, 0, stream>>>(src, src16, 4194304);
  cast_f32_bf16<<<nb(4194304), blk, 0, stream>>>(val, val16, 4194304);
  cast_f32_bf16<<<nb(1048576), blk, 0, stream>>>(Wq, Wq16, 1048576);
  cast_f32_bf16<<<nb(4194304), blk, 0, stream>>>(Wk, Wk16, 4194304);
  cast_f32_bf16<<<nb(4194304), blk, 0, stream>>>(Wv, Wv16, 4194304);
  cast_f32_bf16<<<nb(4194304), blk, 0, stream>>>(Wo, Wo16, 4194304);
  cast_f32_bf16<<<nb(4194304), blk, 0, stream>>>(Wsm, Ws16, 4194304);

  // 2) projections (WMMA GEMMs), 128x128 tiles
  gemm_bf16<<<dim3(HE / 128, 4096 / 128), blk, 0, stream>>>(tgt16, Wq16, bq, Tf32,
                                                            4096, HE, DM);
  gemm_bf16<<<dim3(HE / 128, Sn / 128), blk, 0, stream>>>(src16, Wk16, bk, Kf32,
                                                          Sn, HE, DLLM);
  gemm_bf16<<<dim3(HE / 128, Sn / 128), blk, 0, stream>>>(val16, Wv16, bv, Vtmp,
                                                          Sn, HE, DLLM);

  // 3) head-major bf16 packs (fold score diagonals into T)
  pack_T<<<nb(4194304), blk, 0, stream>>>(Tf32, Ws1, Ws2, Td, 4194304);
  pack_hse<<<nb(1048576), blk, 0, stream>>>(Kf32, Kd, 1048576);
  pack_hse<<<nb(1048576), blk, 0, stream>>>(Vtmp, Vd, 1048576);

  // 4) fused attention
  zero_f32<<<nb(4194304), blk, 0, stream>>>(Amean, 4194304);
  (void)hipFuncSetAttribute((const void*)attn_fused,
                            hipFuncAttributeMaxDynamicSharedMemorySize,
                            (int)ATTN_LDS);
  attn_fused<<<dim3(Ln / 64, Hn, Bn), blk, ATTN_LDS, stream>>>(
      Td, Kd, Vd, Tf32, cw1, cw2, Amean, repb);

  // 5) argmax/argmin over head-mean attention, gather K rows
  argmm_kernel<<<nb((int)NBL), blk, 0, stream>>>(Amean, mxi, mni, (int)NBL);
  gather_kernel<<<dim3((int)NBL), blk, 0, stream>>>(Kf32, mxi, mni, gpos, gneg);

  // 6) output GEMMs -> d_out (out | pos | neg)
  gemm_bf16<<<dim3(DLLM / 128, 4096 / 128), blk, 0, stream>>>(repb, Wo16, bo, outp,
                                                              4096, DLLM, HE);
  gemm_bf16<<<dim3(DLLM / 128, 4096 / 128), blk, 0, stream>>>(
      gpos, Ws16, bsv, outp + (size_t)16777216, 4096, DLLM, HE);
  gemm_bf16<<<dim3(DLLM / 128, 4096 / 128), blk, 0, stream>>>(
      gneg, Ws16, bsv, outp + (size_t)33554432, 4096, DLLM, HE);
}
